// EdgeCondSAGE_3229815407100
// MI455X (gfx1250) — compile-verified, matrix operand order checked by reading, not run on hardware
//
#include <hip/hip_runtime.h>
#include <hip/hip_bf16.h>

typedef __attribute__((ext_vector_type(16))) __bf16 v16bf;
typedef __attribute__((ext_vector_type(8)))  float  v8f;

#define N_NODES 100000
#define N_EDGES 1600000

// ---------------------------------------------------------------------------
// Helpers
// ---------------------------------------------------------------------------
__global__ void zero_f32(float* __restrict__ p, long long n) {
    long long i = (long long)blockIdx.x * blockDim.x + threadIdx.x;
    long long stride = (long long)gridDim.x * blockDim.x;
    for (; i < n; i += stride) p[i] = 0.0f;
}

// Fragment element -> K offset for CDNA5 16-bit 16x32 A layout (B transposed
// analog). Lane half 0 holds K = {0..7, 16..23}, half 1 holds K = {8..15, 24..31}.
// Within a lane, elements 0..7 and 8..15 are each contiguous in K.
__device__ __forceinline__ int frag_k(int e, int half) {
    return (e < 8) ? (half * 8 + e) : (16 + half * 8 + (e - 8));
}

// Swizzle f32 weight [Kreal x 128] row-major into fragment-native bf16 layout:
// dst[((nt*KT + kt)*32 + lane)*16 + e] = W[kt*32 + frag_k(e, lane>>4)][nt*16 + (lane&15)]
// so each lane's v16bf B fragment is one contiguous 32-byte block.
__global__ void cvt_w_sw(const float* __restrict__ src, __bf16* __restrict__ dst,
                         int Kreal, int KT) {
    int i = blockIdx.x * blockDim.x + threadIdx.x;
    int total = 8 * KT * 32 * 16;
    if (i >= total) return;
    int e    = i & 15;
    int lane = (i >> 4) & 31;
    int kt   = (i >> 9) % KT;
    int nt   = i / (KT * 512);
    int k    = kt * 32 + frag_k(e, lane >> 4);
    int col  = nt * 16 + (lane & 15);
    dst[i] = (k < Kreal) ? (__bf16)src[k * 128 + col] : (__bf16)0.0f;
}

// ---------------------------------------------------------------------------
// Message pass: per block 32 edges x 128 outputs (8 waves x [2 M-tiles x 16 N]).
// msg = relu([x[src] | edge_attr] @ W + b); atomic scatter-add into summed[dst].
// ---------------------------------------------------------------------------
__global__ void __launch_bounds__(256)
msg_kernel(const float* __restrict__ x,        // [N,128] node features
           const int*   __restrict__ ei,       // [2,E]
           const float* __restrict__ ea,       // [E,16]
           const __bf16* __restrict__ Wsw,     // swizzled [8][5][32][16] bf16
           const float* __restrict__ bmsg,     // [128]
           float* __restrict__ summed,         // [N,128]
           float* __restrict__ cnt)            // [N]
{
    __shared__ __bf16 As[32 * 160];
    __shared__ int sSrc[32];
    __shared__ int sDst[32];

    const int tid = threadIdx.x;
    const int e0  = blockIdx.x * 32;

    if (tid < 32) {
        sSrc[tid] = ei[e0 + tid];
        sDst[tid] = ei[N_EDGES + e0 + tid];
    }
    __syncthreads();

    // Stage A tile [32 x 160] bf16: x[src] | edge_attr | zero pad (144->160).
    for (int i = tid; i < 32 * 160; i += 256) {
        int r = i / 160;
        int k = i - r * 160;
        float v;
        if (k < 128)      v = x[(size_t)sSrc[r] * 128 + k];
        else if (k < 144) v = ea[(size_t)(e0 + r) * 16 + (k - 128)];
        else              v = 0.0f;
        As[i] = (__bf16)v;
    }
    __syncthreads();

    const int wave  = tid >> 5;
    const int lane  = tid & 31;
    const int half  = lane >> 4;
    const int mn    = lane & 15;
    const int nbase = wave * 16;

    v8f c0 = {0.f,0.f,0.f,0.f,0.f,0.f,0.f,0.f};
    v8f c1 = {0.f,0.f,0.f,0.f,0.f,0.f,0.f,0.f};
#pragma unroll
    for (int kt = 0; kt < 5; ++kt) {
        const int k0 = kt * 32;
        // One contiguous 32B load per lane for the whole B fragment.
        v16bf b = *(const v16bf*)&Wsw[(((size_t)wave * 5 + kt) * 32 + lane) * 16];
        v16bf a0, a1;
#pragma unroll
        for (int e = 0; e < 16; ++e) {
            int K = frag_k(e, half);
            a0[e] = As[mn * 160 + k0 + K];          // edges 0..15
            a1[e] = As[(mn + 16) * 160 + k0 + K];   // edges 16..31
        }
        c0 = __builtin_amdgcn_wmma_f32_16x16x32_bf16(
                 false, a0, false, b, (short)0, c0, false, false);
        c1 = __builtin_amdgcn_wmma_f32_16x16x32_bf16(
                 false, a1, false, b, (short)0, c1, false, false);
    }

    const int   col = nbase + mn;
    const float bia = bmsg[col];
#pragma unroll
    for (int r = 0; r < 8; ++r) {
        int   m  = r + 8 * half;
        float v0 = c0[r] + bia;
        float v1 = c1[r] + bia;
        v0 = v0 > 0.0f ? v0 : 0.0f;
        v1 = v1 > 0.0f ? v1 : 0.0f;
        atomicAdd(&summed[(size_t)sDst[m] * 128 + col], v0);
        atomicAdd(&summed[(size_t)sDst[m + 16] * 128 + col], v1);
    }
    if (tid < 32) atomicAdd(&cnt[sDst[tid]], 1.0f);
}

// ---------------------------------------------------------------------------
// Update: per block 32 nodes x 128 outputs.
// out = normalize([x | summed/max(cnt,1)] @ W + b_upd) + bias  (opt. ReLU)
// ---------------------------------------------------------------------------
__global__ void __launch_bounds__(256)
upd_kernel(const float* __restrict__ xin,      // [N,128] layer input
           const float* __restrict__ summed,   // [N,128]
           const float* __restrict__ cnt,      // [N]
           const __bf16* __restrict__ Wsw,     // swizzled [8][8][32][16] bf16
           const float* __restrict__ bupd,     // [128]
           const float* __restrict__ bias,     // [128]
           float* __restrict__ out,            // [N,128] (row-local alias of xin ok)
           int applyRelu)
{
    __shared__ __bf16 As[32 * 256];   // 16 KB
    __shared__ float invc[32];
    __shared__ float nrm2[32];

    const int tid = threadIdx.x;
    const int n0  = blockIdx.x * 32;

    if (tid < 32) {
        invc[tid] = 1.0f / fmaxf(cnt[n0 + tid], 1.0f);
        nrm2[tid] = 0.0f;
    }
    __syncthreads();

    // Stage A tile [32 x 256] bf16: x | mean-aggregated messages.
    for (int i = tid; i < 32 * 256; i += 256) {
        int r = i >> 8;
        int k = i & 255;
        float v = (k < 128)
                    ? xin[(size_t)(n0 + r) * 128 + k]
                    : summed[(size_t)(n0 + r) * 128 + (k - 128)] * invc[r];
        As[i] = (__bf16)v;
    }
    __syncthreads();

    const int wave  = tid >> 5;
    const int lane  = tid & 31;
    const int half  = lane >> 4;
    const int mn    = lane & 15;
    const int nbase = wave * 16;

    v8f c0 = {0.f,0.f,0.f,0.f,0.f,0.f,0.f,0.f};
    v8f c1 = {0.f,0.f,0.f,0.f,0.f,0.f,0.f,0.f};
#pragma unroll
    for (int kt = 0; kt < 8; ++kt) {
        const int k0 = kt * 32;
        v16bf b = *(const v16bf*)&Wsw[(((size_t)wave * 8 + kt) * 32 + lane) * 16];
        v16bf a0, a1;
#pragma unroll
        for (int e = 0; e < 16; ++e) {
            int K = frag_k(e, half);
            a0[e] = As[mn * 256 + k0 + K];
            a1[e] = As[(mn + 16) * 256 + k0 + K];
        }
        c0 = __builtin_amdgcn_wmma_f32_16x16x32_bf16(
                 false, a0, false, b, (short)0, c0, false, false);
        c1 = __builtin_amdgcn_wmma_f32_16x16x32_bf16(
                 false, a1, false, b, (short)0, c1, false, false);
    }

    const int   col = nbase + mn;
    const float bu  = bupd[col];
    float vals0[8], vals1[8];
#pragma unroll
    for (int r = 0; r < 8; ++r) {
        int m    = r + 8 * half;
        vals0[r] = c0[r] + bu;
        vals1[r] = c1[r] + bu;
        atomicAdd(&nrm2[m],      vals0[r] * vals0[r]);   // ds_add_f32 row reduce
        atomicAdd(&nrm2[m + 16], vals1[r] * vals1[r]);
    }
    __syncthreads();

    const float bo = bias[col];
#pragma unroll
    for (int r = 0; r < 8; ++r) {
        int   m    = r + 8 * half;
        float inv0 = 1.0f / fmaxf(sqrtf(nrm2[m]), 1e-12f);
        float inv1 = 1.0f / fmaxf(sqrtf(nrm2[m + 16]), 1e-12f);
        float o0   = vals0[r] * inv0 + bo;
        float o1   = vals1[r] * inv1 + bo;
        if (applyRelu) { o0 = o0 > 0.f ? o0 : 0.f; o1 = o1 > 0.f ? o1 : 0.f; }
        out[(size_t)(n0 + m) * 128 + col]      = o0;
        out[(size_t)(n0 + m + 16) * 128 + col] = o1;
    }
}

// ---------------------------------------------------------------------------
extern "C" void kernel_launch(void* const* d_in, const int* in_sizes, int n_in,
                              void* d_out, int out_size, void* d_ws, size_t ws_size,
                              hipStream_t stream) {
    const float* x   = (const float*)d_in[0];
    const int*   ei  = (const int*)  d_in[1];   // [2,E] (JAX x64-off -> int32)
    const float* ea  = (const float*)d_in[2];
    const float* wm1 = (const float*)d_in[3];
    const float* bm1 = (const float*)d_in[4];
    const float* wu1 = (const float*)d_in[5];
    const float* bu1 = (const float*)d_in[6];
    const float* bi1 = (const float*)d_in[7];
    const float* wm2 = (const float*)d_in[8];
    const float* bm2 = (const float*)d_in[9];
    const float* wu2 = (const float*)d_in[10];
    const float* bu2 = (const float*)d_in[11];
    const float* bi2 = (const float*)d_in[12];
    float* out = (float*)d_out;

    char* ws = (char*)d_ws;
    float*  summed = (float*)ws;                                       // N*128 f32
    float*  cnt    = (float*)(ws + (size_t)N_NODES * 128 * 4);         // N f32
    __bf16* Wm1    = (__bf16*)(ws + (size_t)N_NODES * 128 * 4 + (size_t)N_NODES * 4);
    __bf16* Wu1    = Wm1 + 8 * 5 * 32 * 16;   // msg: 5 k-tiles (K=144 pad 160)
    __bf16* Wm2    = Wu1 + 8 * 8 * 32 * 16;   // upd: 8 k-tiles (K=256)
    __bf16* Wu2    = Wm2 + 8 * 5 * 32 * 16;

    const int TM = 8 * 5 * 32 * 16;   // 20480 elems
    const int TU = 8 * 8 * 32 * 16;   // 32768 elems
    cvt_w_sw<<<(TM + 255) / 256, 256, 0, stream>>>(wm1, Wm1, 144, 5);
    cvt_w_sw<<<(TU + 255) / 256, 256, 0, stream>>>(wu1, Wu1, 256, 8);
    cvt_w_sw<<<(TM + 255) / 256, 256, 0, stream>>>(wm2, Wm2, 144, 5);
    cvt_w_sw<<<(TU + 255) / 256, 256, 0, stream>>>(wu2, Wu2, 256, 8);

    const long long zn = (long long)N_NODES * 128 + N_NODES;  // summed+cnt contiguous

    // ---- Layer 1 ----
    zero_f32<<<2048, 256, 0, stream>>>(summed, zn);
    msg_kernel<<<N_EDGES / 32, 256, 0, stream>>>(x, ei, ea, Wm1, bm1, summed, cnt);
    upd_kernel<<<N_NODES / 32, 256, 0, stream>>>(x, summed, cnt, Wu1, bu1, bi1,
                                                 out, /*relu=*/1);
    // ---- Layer 2 (reads h from d_out, overwrites d_out row-locally) ----
    zero_f32<<<2048, 256, 0, stream>>>(summed, zn);
    msg_kernel<<<N_EDGES / 32, 256, 0, stream>>>(out, ei, ea, Wm2, bm2, summed, cnt);
    upd_kernel<<<N_NODES / 32, 256, 0, stream>>>(out, summed, cnt, Wu2, bu2, bi2,
                                                 out, /*relu=*/0);
}
